// GeometricContrastiveLoss_26173530701949
// MI455X (gfx1250) — compile-verified, matrix operand order checked by reading, not run on hardware
//
#include <hip/hip_runtime.h>
#include <cstdint>

typedef __attribute__((ext_vector_type(16))) _Float16 v16h;
typedef __attribute__((ext_vector_type(8)))  float    v8f;
typedef __attribute__((ext_vector_type(4)))  unsigned int u32x4;
typedef __attribute__((ext_vector_type(8)))  int      i32x8;
typedef __attribute__((ext_vector_type(4)))  int      i32x4;

// Problem dims (fixed by setup_inputs)
constexpr int B  = 8;
constexpr int NV = 4;
constexpr int D  = 256;
constexpr int H  = 48;
constexpr int W  = 64;
constexpr int HW = H * W;              // 3072
constexpr int NPAIR = NV - 1;          // 3
constexpr int NPB = NPAIR * B;         // 24 pair-batches
constexpr int NA = 256;                // anchors per pair-batch
constexpr float INV_T   = 1.0f / 0.07f;
constexpr float THRESH2 = 0.1f * 0.1f;

// ---- TDM availability / arity detection (compile-safe hedging) ------------
#if defined(__has_builtin)
#if __has_builtin(__builtin_amdgcn_tensor_load_to_lds) && \
    __has_builtin(__builtin_amdgcn_s_wait_tensorcnt)
#define USE_TDM 1
#else
#define USE_TDM 0
#endif
#else
#define USE_TDM 0
#endif

#if defined(__has_include)
#if __has_include(<hip/amd_detail/amd_gfx1250_TDM.h>)
#define TDM_6ARG 1
#else
#define TDM_6ARG 0
#endif
#else
#define TDM_6ARG 0
#endif

#if USE_TDM
// 1-D TDM copy of 8KB (1024 x 8-byte elements) global -> LDS.
// D# per CDNA5 ISA ch.8: group0 = {count=1, lds_addr, global_addr, type=2};
// group1 = {data_size=3 (8B), tensor_dim0=1024, tile_dim0=1024, stride=1024}.
static __device__ __forceinline__ void tdm_load_8k(unsigned ldsOff,
                                                   unsigned long long ga) {
  u32x4 g0 = {1u,                                   // count=1, user mode
              ldsOff,                               // lds_addr (bytes)
              (unsigned)(ga & 0xFFFFFFFFu),         // global_addr[31:0]
              (unsigned)((ga >> 32) & 0x1FFFFFFu) | (2u << 30)}; // [56:32]|type=2
  i32x8 g1 = {0x00030000,                           // wg_mask=0, data_size=3 (8B)
              (int)(1024u << 16),                   // tensor_dim0[15:0] @ bits63:48
              0,                                    // tensor_dim0 hi / tensor_dim1 lo
              (int)(1024u << 16),                   // tile_dim0 @ bits127:112
              0,                                    // tile_dim1=0, tile_dim2=0 (1-D)
              1024,                                 // tensor_dim0_stride lo32
              0, 0};
  i32x4 gz = {0, 0, 0, 0};
#if TDM_6ARG
  i32x8 gz8 = {0, 0, 0, 0, 0, 0, 0, 0};
  __builtin_amdgcn_tensor_load_to_lds(g0, g1, gz, gz, gz8, 0);
#else
  __builtin_amdgcn_tensor_load_to_lds(g0, g1, gz, gz, 0);
#endif
}
#endif

// ---------------------------------------------------------------------------
// K1a: per-point reciprocal L2 norms of features + point validity + counts
// ---------------------------------------------------------------------------
__global__ void norms_valid_kernel(const float* __restrict__ feat,
                                   const float* __restrict__ pts,
                                   float* __restrict__ rnorm,
                                   unsigned char* __restrict__ valid,
                                   int* __restrict__ vcount) {
  int p  = blockIdx.x * blockDim.x + threadIdx.x;   // < B*NV*HW
  int bn = p / HW;
  int hw = p - bn * HW;
  const float* f = feat + (size_t)bn * D * HW + hw;
  float ss = 0.f;
#pragma unroll 4
  for (int d = 0; d < D; ++d) {
    float x = f[(size_t)d * HW];
    ss += x * x;
  }
  float nrm = sqrtf(ss);
  rnorm[p] = 1.0f / fmaxf(nrm, 1e-12f);

  const float* q = pts + (size_t)p * 3;
  float s2 = q[0] * q[0] + q[1] * q[1] + q[2] * q[2];
  bool v = s2 > 1e-12f;                 // norm > 1e-6
  valid[p] = v ? 1 : 0;
  unsigned long long m = __ballot(v);
  if ((threadIdx.x & 31) == 0) atomicAdd(&vcount[bn], (int)__popcll(m));
}

// ---------------------------------------------------------------------------
// K1b: normalize + transpose to f16 [B,N,HW,D] via 64x64 LDS tile
// ---------------------------------------------------------------------------
__global__ void normalize_f16_kernel(const float* __restrict__ feat,
                                     const float* __restrict__ rnorm,
                                     _Float16* __restrict__ nf) {
  __shared__ float tile[64][65];
  int bn  = blockIdx.z;
  int hw0 = blockIdx.x * 64;
  int d0  = blockIdx.y * 64;
  int t   = threadIdx.x;                // 256 threads

  const float* src = feat + ((size_t)bn * D + d0) * HW + hw0;
  int hwl = t & 63;
  int dl0 = t >> 6;                     // 0..3
#pragma unroll
  for (int i = 0; i < 16; ++i) {
    int dl = dl0 + i * 4;
    tile[dl][hwl] = src[(size_t)dl * HW + hwl];   // coalesced along hw
  }
  __syncthreads();

  int dp = t & 31;                      // d-pair index 0..31
  int hb = t >> 5;                      // 0..7
#pragma unroll
  for (int i = 0; i < 8; ++i) {
    int hwl2 = hb + i * 8;
    float rn = rnorm[(size_t)bn * HW + hw0 + hwl2];
    union { _Float16 h[2]; unsigned u; } pk;
    pk.h[0] = (_Float16)(tile[dp * 2 + 0][hwl2] * rn);
    pk.h[1] = (_Float16)(tile[dp * 2 + 1][hwl2] * rn);
    unsigned* dst = (unsigned*)(nf + ((size_t)bn * HW + hw0 + hwl2) * D + d0);
    dst[dp] = pk.u;                     // coalesced half2 stores
  }
}

// ---------------------------------------------------------------------------
// K2: one wave per anchor: nearest neighbor in 3D + anchor validity
// ---------------------------------------------------------------------------
__global__ void anchor_nn_kernel(const float* __restrict__ pts,
                                 const unsigned char* __restrict__ valid,
                                 const int* __restrict__ aidx,
                                 int* __restrict__ pos_idx,
                                 unsigned char* __restrict__ validA,
                                 int* __restrict__ cntA) {
  int wid  = (blockIdx.x * blockDim.x + threadIdx.x) >> 5;
  int lane = threadIdx.x & 31;
  int pb = wid >> 8;                    // 0..23
  int a  = wid & 255;
  int v  = pb / B + 1;
  int b  = pb - (v - 1) * B;
  int ahw = aidx[((v - 1) * B + b) * NA + a];

  const float* pi = pts + (((size_t)b * NV + 0) * HW + ahw) * 3;
  float ax = pi[0], ay = pi[1], az = pi[2];
  const float* pj = pts + (((size_t)b * NV + v) * HW) * 3;
  const unsigned char* vj = valid + ((size_t)b * NV + v) * HW;

  float best = 3.0e38f; int bj = 0;
  for (int j = lane; j < HW; j += 32) {
    float dx = ax - pj[j * 3 + 0];
    float dy = ay - pj[j * 3 + 1];
    float dz = az - pj[j * 3 + 2];
    float d2 = dx * dx + dy * dy + dz * dz;
    if (!vj[j]) d2 = 3.0e38f;
    if (d2 < best) { best = d2; bj = j; }
  }
#pragma unroll
  for (int off = 16; off > 0; off >>= 1) {
    float ob = __shfl_xor(best, off, 32);
    int   oj = __shfl_xor(bj, off, 32);
    if (ob < best || (ob == best && oj < bj)) { best = ob; bj = oj; }
  }
  if (lane == 0) {
    int ok = (best < THRESH2) && valid[((size_t)b * NV + 0) * HW + ahw];
    pos_idx[pb * NA + a] = bj;
    validA[pb * NA + a]  = (unsigned char)ok;
    if (ok) atomicAdd(&cntA[pb], 1);
  }
}

// ---------------------------------------------------------------------------
// K3: one wave per anchor: positive similarity (dot of normalized f16 rows)
// ---------------------------------------------------------------------------
__global__ void pos_sim_kernel(const _Float16* __restrict__ nf,
                               const int* __restrict__ aidx,
                               const int* __restrict__ pos_idx,
                               float* __restrict__ pos_sim) {
  int wid  = (blockIdx.x * blockDim.x + threadIdx.x) >> 5;
  int lane = threadIdx.x & 31;
  int pb = wid >> 8, a = wid & 255;
  int v  = pb / B + 1;
  int b  = pb - (v - 1) * B;
  int ahw = aidx[((v - 1) * B + b) * NA + a];
  int pj  = pos_idx[pb * NA + a];
  const _Float16* ra = nf + (((size_t)b * NV + 0) * HW + ahw) * D + lane * 8;
  const _Float16* rp = nf + (((size_t)b * NV + v) * HW + pj) * D + lane * 8;
  float s = 0.f;
#pragma unroll
  for (int i = 0; i < 8; ++i) s += (float)ra[i] * (float)rp[i];
#pragma unroll
  for (int off = 16; off > 0; off >>= 1) s += __shfl_xor(s, off, 32);
  if (lane == 0) pos_sim[pb * NA + a] = s * INV_T;
}

// ---------------------------------------------------------------------------
// K4: WMMA similarity GEMM fused with masked online logsumexp.
// grid = (24 pair-batches, 8 anchor groups of 32), block = 128 (4 waves).
// M-tile = 32 anchors: two A fragment sets share every B fragment.
// TDM double-buffered pipeline: each wave's next 16-column B tile (one
// contiguous 8KB block of nf) is DMA'd to LDS by tensor_load_to_lds while the
// current tile's 16 WMMAs execute; synchronized with s_wait_tensorcnt.
// ---------------------------------------------------------------------------
__global__ void __launch_bounds__(128)
sim_lse_kernel(const _Float16* __restrict__ nf,
               const int* __restrict__ aidx,
               const unsigned char* __restrict__ valid,
               const unsigned char* __restrict__ validA,
               const float* __restrict__ pos_sim,
               float* __restrict__ loss_sum) {
  // [0, 8192)            : As  (32 anchor rows x 256 halves, 16KB)
  // [8192, 8192+32768)   : B double buffers: wave w -> 2 x 4096 halves (8KB each)
  __shared__ _Float16 lds[32 * D + 4 * 2 * 16 * D];
  __shared__ float Mp[4][32];
  __shared__ float Sp[4][32];
  _Float16* As = lds;

  int pb = blockIdx.x;                  // 0..23
  int g  = blockIdx.y;                  // 0..7  (group of 32 anchors)
  int v  = pb / B + 1;
  int b  = pb - (v - 1) * B;
  int t = threadIdx.x, wave = t >> 5, lane = t & 31;

  // cooperative load of 32 gathered anchor rows (each 512B contiguous)
  {
    int row = t >> 2;                   // 0..31
    int seg = t & 3;                    // 0..3 (128B each)
    int ahw = aidx[((v - 1) * B + b) * NA + g * 32 + row];
    const uint4* src = (const uint4*)(nf + (((size_t)b * NV + 0) * HW + ahw) * D) + seg * 8;
    uint4* dst = (uint4*)(As + row * D) + seg * 8;
#pragma unroll
    for (int i = 0; i < 8; ++i) dst[i] = src[i];
  }
  __syncthreads();

  // Prebuild 2x8 A fragments (ISA 16-bit A 16x32 lane layout:
  // lanes 0-15 hold K {0..7,16..23}, lanes 16-31 hold K {8..15,24..31})
  int lm    = lane & 15;
  int khalf = (lane >> 4) << 3;         // 0 or 8
  v16h afrag0[8], afrag1[8];
#pragma unroll
  for (int k = 0; k < 8; ++k) {
    const _Float16* a0 = As + lm * D + k * 32 + khalf;
    const _Float16* a1 = a0 + 16 * D;
#pragma unroll
    for (int i = 0; i < 8; ++i) {
      afrag0[k][i] = a0[i]; afrag0[k][8 + i] = a0[16 + i];
      afrag1[k][i] = a1[i]; afrag1[k][8 + i] = a1[16 + i];
    }
  }

  const _Float16* nfj = nf + (((size_t)b * NV + v) * HW) * D;
  const unsigned char* vj = valid + ((size_t)b * NV + v) * HW;

  float mr[16], sr[16];
#pragma unroll
  for (int r = 0; r < 16; ++r) { mr[r] = -1.0e30f; sr[r] = 0.f; }

  constexpr int NT = HW / 64;           // 48 column tiles per wave
  _Float16* bbuf0 = lds + 32 * D + wave * 2 * 16 * D;
  _Float16* bbuf1 = bbuf0 + 16 * D;

#if USE_TDM
  unsigned bOff0 = (unsigned)(uintptr_t)bbuf0;  // LDS aperture: addr[31:0] = offset
  unsigned bOff1 = (unsigned)(uintptr_t)bbuf1;
  // prime the pipe: tile 0 -> buffer 0
  tdm_load_8k(bOff0, (unsigned long long)(uintptr_t)(nfj + (size_t)(wave * 16) * D));
#endif

  for (int it = 0; it < NT; ++it) {
    int j0 = wave * 16 + it * 64;
    int j  = j0 + lm;
    const _Float16* bsrc;

#if USE_TDM
    if (it + 1 < NT) {
      int jn = wave * 16 + (it + 1) * 64;
      tdm_load_8k((it & 1) ? bOff0 : bOff1,
                  (unsigned long long)(uintptr_t)(nfj + (size_t)jn * D));
      __builtin_amdgcn_s_wait_tensorcnt(1);   // current tile's DMA complete
    } else {
      __builtin_amdgcn_s_wait_tensorcnt(0);
    }
    bsrc = ((it & 1) ? bbuf1 : bbuf0) + lm * D;
#else
    bsrc = nfj + (size_t)j * D;
    __builtin_prefetch(nfj + (size_t)(j + 64) * D, 0, 3);
#endif

    v8f acc0, acc1;
#pragma unroll
    for (int r = 0; r < 8; ++r) { acc0[r] = 0.f; acc1[r] = 0.f; }

#pragma unroll
    for (int k = 0; k < 8; ++k) {
      const _Float16* bp = bsrc + k * 32 + khalf;
      v16h bfrag;
#pragma unroll
      for (int i = 0; i < 8; ++i) { bfrag[i] = bp[i]; bfrag[8 + i] = bp[16 + i]; }
      acc0 = __builtin_amdgcn_wmma_f32_16x16x32_f16(false, afrag0[k], false, bfrag,
                                                    (short)0, acc0, false, false);
      acc1 = __builtin_amdgcn_wmma_f32_16x16x32_f16(false, afrag1[k], false, bfrag,
                                                    (short)0, acc1, false, false);
    }

    // D layout: lane (mod 16) = column j; VGPR r = row r (+8 for upper half)
    if (vj[j]) {
#pragma unroll
      for (int r = 0; r < 8; ++r) {
        float v0 = acc0[r] * INV_T;
        float n0 = fmaxf(mr[r], v0);
        sr[r] = sr[r] * __expf(mr[r] - n0) + __expf(v0 - n0);
        mr[r] = n0;
        float v1 = acc1[r] * INV_T;
        float n1 = fmaxf(mr[8 + r], v1);
        sr[8 + r] = sr[8 + r] * __expf(mr[8 + r] - n1) + __expf(v1 - n1);
        mr[8 + r] = n1;
      }
    }
  }

  // combine the 16 lanes of each half (they share the same rows)
#pragma unroll
  for (int off = 1; off < 16; off <<= 1) {
#pragma unroll
    for (int r = 0; r < 16; ++r) {
      float om = __shfl_xor(mr[r], off, 32);
      float os = __shfl_xor(sr[r], off, 32);
      float nm = fmaxf(mr[r], om);
      sr[r] = sr[r] * __expf(mr[r] - nm) + os * __expf(om - nm);
      mr[r] = nm;
    }
  }
  if (lane == 0 || lane == 16) {
    int base = (lane >> 4) * 8;          // lower half: rows 0-7, upper: 8-15
#pragma unroll
    for (int r = 0; r < 8; ++r) {
      Mp[wave][base + r]      = mr[r];       // anchors 0..15 of the group
      Sp[wave][base + r]      = sr[r];
      Mp[wave][16 + base + r] = mr[8 + r];   // anchors 16..31 of the group
      Sp[wave][16 + base + r] = sr[8 + r];
    }
  }
  __syncthreads();

  if (t < 32) {
    float M = Mp[0][t], S = Sp[0][t];
#pragma unroll
    for (int w = 1; w < 4; ++w) {
      float om = Mp[w][t], os = Sp[w][t];
      float nm = fmaxf(M, om);
      S = S * __expf(M - nm) + os * __expf(om - nm);
      M = nm;
    }
    int idx = pb * NA + g * 32 + t;
    if (validA[idx]) {
      float lse = M + logf(S);
      atomicAdd(&loss_sum[pb], lse - pos_sim[idx]);
    }
  }
}

// ---------------------------------------------------------------------------
// K5: finalize the masked means into the scalar loss
// ---------------------------------------------------------------------------
__global__ void finalize_kernel(const float* __restrict__ loss_sum,
                                const int* __restrict__ cntA,
                                const int* __restrict__ vcount,
                                float* __restrict__ out) {
  if (threadIdx.x != 0 || blockIdx.x != 0) return;
  float total = 0.f;
  for (int v = 1; v < NV; ++v) {
    float psum = 0.f; int ninc = 0;
    for (int b = 0; b < B; ++b) {
      int pb = (v - 1) * B + b;
      int c  = cntA[pb];
      int cd = c > 1 ? c : 1;
      float lb = loss_sum[pb] / (float)cd;
      bool inc = (c >= 5) && (vcount[b * NV + 0] >= 10) && (vcount[b * NV + v] >= 10);
      if (inc) { psum += lb; ninc++; }
    }
    total += (ninc > 0) ? psum / (float)ninc : 0.f;
  }
  out[0] = total / (float)(NV - 1);
}

// ---------------------------------------------------------------------------
extern "C" void kernel_launch(void* const* d_in, const int* in_sizes, int n_in,
                              void* d_out, int out_size, void* d_ws, size_t ws_size,
                              hipStream_t stream) {
  const float* feat = (const float*)d_in[0];   // [B,N,D,H,W] fp32
  const float* pts  = (const float*)d_in[1];   // [B,N,H,W,3] fp32
  const int*   aidx = (const int*)d_in[2];     // [N-1,B,256] int32

  char* ws = (char*)d_ws;
  size_t off = 0;
  auto take = [&](size_t bytes) -> void* {
    off = (off + 255) & ~size_t(255);
    void* p = ws + off;
    off += bytes;
    return p;
  };

  _Float16* nf   = (_Float16*)take((size_t)B * NV * HW * D * 2);   // 50.3 MB
  float* rnorm   = (float*)take((size_t)B * NV * HW * 4);
  unsigned char* valid  = (unsigned char*)take((size_t)B * NV * HW);
  int* pos_idx   = (int*)take((size_t)NPB * NA * 4);
  unsigned char* validA = (unsigned char*)take((size_t)NPB * NA);
  float* pos_sim = (float*)take((size_t)NPB * NA * 4);
  // zero-initialized accumulators (contiguous for a single memset)
  float* loss_sum = (float*)take(NPB * 4 + NPB * 4 + B * NV * 4);
  int* cntA   = (int*)(loss_sum + NPB);
  int* vcount = cntA + NPB;

  hipMemsetAsync(loss_sum, 0, NPB * 4 + NPB * 4 + B * NV * 4, stream);

  norms_valid_kernel<<<(B * NV * HW) / 256, 256, 0, stream>>>(
      feat, pts, rnorm, valid, vcount);

  dim3 g1(HW / 64, D / 64, B * NV);
  normalize_f16_kernel<<<g1, 256, 0, stream>>>(feat, rnorm, nf);

  anchor_nn_kernel<<<(NPB * NA) / 4, 128, 0, stream>>>(
      pts, valid, aidx, pos_idx, validA, cntA);

  pos_sim_kernel<<<(NPB * NA) / 4, 128, 0, stream>>>(
      nf, aidx, pos_idx, pos_sim);

  dim3 g4(NPB, NA / 32);
  sim_lse_kernel<<<g4, 128, 0, stream>>>(
      nf, aidx, valid, validA, pos_sim, loss_sum);

  finalize_kernel<<<1, 32, 0, stream>>>(loss_sum, cntA, vcount, (float*)d_out);
}
